// RelationNetwork_12893491823127
// MI455X (gfx1250) — matrix operand order repaired and checked by reading, not compiled, not run on hardware
//
#include <hip/hip_runtime.h>
#include <hip/hip_bf16.h>

typedef __attribute__((ext_vector_type(16))) _Float16 v16h;
typedef __attribute__((ext_vector_type(8)))  _Float16 v8h;
typedef __attribute__((ext_vector_type(8)))  float    v8f;

#define TOK   1024
#define DM    128
#define NHEADS 4
#define HDIM  32
#define FFD   2048
#define LN_EPS 1e-5f

// ---------------------------------------------------------------------------
// gfx1250 async copy: global -> LDS, 16 bytes per lane (ASYNCcnt-tracked)
// ---------------------------------------------------------------------------
__device__ __forceinline__ void async_b128(void* lds, const _Float16* g)
{
    asm volatile("global_load_async_to_lds_b128 %0, %1, off"
                 :: "v"((unsigned int)(uintptr_t)lds),
                    "v"((unsigned long long)(uintptr_t)g)
                 : "memory");
}
__device__ __forceinline__ void wait_async_le2()
{
    asm volatile("s_wait_asynccnt 0x2" ::: "memory");
}
__device__ __forceinline__ void wait_async_0()
{
    asm volatile("s_wait_asynccnt 0x0" ::: "memory");
}

// ---------------------------------------------------------------------------
// WMMA GEMM:  C[M,N] = alpha * (A[M,K] @ B[N,K]^T) + bias[N]  (opt ReLU)
// A,B f16 row-major (lda/ldb); optional f32/f16 outputs; grid.z = heads with
// per-head element offsets sA/sB/sC. Block = 128 threads (4 waves) computing
// a 64 x (16*NT) tile of C; the 64x32 A slab per K-step is staged into LDS
// with double-buffered global_load_async_to_lds_b128.
// ---------------------------------------------------------------------------
template<int NT>
__global__ void gemm_wmma_nt(const _Float16* __restrict__ A, int lda, long sA,
                             const _Float16* __restrict__ B, int ldb, long sB,
                             const float* __restrict__ bias,
                             float* __restrict__ C32, _Float16* __restrict__ C16,
                             int ldc, long sC,
                             int M, int N, int K, float alpha, int relu)
{
    __shared__ _Float16 atile[2][64][32];

    const int head = blockIdx.z;
    A += (size_t)head * sA;
    B += (size_t)head * sB;
    if (C32) C32 += (size_t)head * sC;
    if (C16) C16 += (size_t)head * sC;

    const int tid  = threadIdx.x;
    const int lane = tid & 31;
    const int wave = tid >> 5;
    const int mblk  = blockIdx.y * 64;
    const int nbase = blockIdx.x * (16 * NT);

    // async staging: 64x32 halves = 4KB = 256 x b128; 2 chunks per thread
    const int ch  = tid * 2;
    const int sr0 = ch >> 2,        sc0 = (ch & 3) * 8;
    const int sr1 = (ch + 1) >> 2,  sc1 = ((ch + 1) & 3) * 8;

    // A fragment addressing (16x32 f16, contiguous k = 16h + s), read from LDS
    const int mA  = lane & 15;
    const int kbA = (lane >> 4) * 16;
    // B fragment: lane holds one output column, 16 contiguous K-halves (32B)
    const int nB  = lane & 15;
    const int kbB = (lane >> 4) * 16;

    v8f acc[NT] = {};

    const int ksteps = K / 32;

    // prologue: stage K-step 0 into buffer 0
    async_b128(&atile[0][sr0][sc0], A + (size_t)(mblk + sr0) * lda + sc0);
    async_b128(&atile[0][sr1][sc1], A + (size_t)(mblk + sr1) * lda + sc1);

    for (int i = 0; i < ksteps; ++i) {
        const int k0 = i * 32;
        if (i + 1 < ksteps) {
            const int kn = k0 + 32;
            _Float16 (*buf)[32] = atile[(i + 1) & 1];
            async_b128(&buf[sr0][sc0], A + (size_t)(mblk + sr0) * lda + kn + sc0);
            async_b128(&buf[sr1][sc1], A + (size_t)(mblk + sr1) * lda + kn + sc1);
            wait_async_le2();            // current buffer (first 2 ops) landed
        } else {
            wait_async_0();
        }
        __syncthreads();                 // other waves' async data visible

        const _Float16* la = &atile[i & 1][wave * 16 + mA][0];
        v16h af;
        {
            v8h lo = *(const v8h*)(la + kbA);        // ds_load_b128
            v8h hi = *(const v8h*)(la + 8 + kbA);
            #pragma unroll
            for (int q = 0; q < 8; ++q) { af[q] = lo[q]; af[q + 8] = hi[q]; }
        }

        __builtin_prefetch(B + (size_t)(nbase + nB) * ldb + k0 + 32, 0, 1);
        #pragma unroll
        for (int t = 0; t < NT; ++t) {
            const _Float16* pb = B + (size_t)(nbase + t * 16 + nB) * ldb + k0 + kbB;
            v16h bf = *(const v16h*)pb;
            acc[t] = __builtin_amdgcn_wmma_f32_16x16x32_f16(
                false, af, false, bf, (short)0, acc[t], false, false);
        }
        __syncthreads();                 // protect buffer reuse next iteration
    }

    // C/D layout: VGPR r holds row r (lanes 0-15) / row r+8 (lanes 16-31)
    const int mtile  = mblk + wave * 16;
    const int rowoff = (lane >> 4) * 8;
    const int coloff = lane & 15;
    #pragma unroll
    for (int t = 0; t < NT; ++t) {
        const int col = nbase + t * 16 + coloff;
        const float bv = bias ? bias[col] : 0.0f;
        #pragma unroll
        for (int r = 0; r < 8; ++r) {
            const int row = mtile + rowoff + r;
            float v = acc[t][r] * alpha + bv;
            if (relu) v = fmaxf(v, 0.0f);
            const size_t idx = (size_t)row * ldc + col;
            if (C32) C32[idx] = v;
            if (C16) C16[idx] = (_Float16)v;
        }
    }
}

// ---------------------------------------------------------------------------
__global__ void embed_kernel(const int* __restrict__ ids,
                             const float* __restrict__ emb,
                             float* __restrict__ x, _Float16* __restrict__ xh)
{
    const int n = blockIdx.x, t = threadIdx.x;
    const float v = emb[(size_t)ids[n] * DM + t];
    x[n * DM + t]  = v;
    xh[n * DM + t] = (_Float16)v;
}

__global__ void f32_to_f16_kernel(const float* __restrict__ s,
                                  _Float16* __restrict__ d, int n)
{
    const int i = blockIdx.x * 256 + threadIdx.x;
    if (i < n) d[i] = (_Float16)s[i];
}

// vt[h][c][n] = qkv[n][2D + h*32 + c]   (transposed V for the PV GEMM)
__global__ void build_vt_kernel(const _Float16* __restrict__ qkv,
                                _Float16* __restrict__ vt)
{
    const int idx = blockIdx.x * 256 + threadIdx.x;   // 4*32*1024 total
    const int n = idx & (TOK - 1);
    const int c = (idx >> 10) & (HDIM - 1);
    const int h = idx >> 15;
    vt[idx] = qkv[(size_t)n * (3 * DM) + 2 * DM + h * HDIM + c];
}

// one block (128 thr) per attention row; 1024 elements per row
__global__ void softmax_kernel(const float* __restrict__ S,
                               _Float16* __restrict__ P)
{
    const size_t row = blockIdx.x;
    const float* s = S + row * TOK;
    _Float16*    p = P + row * TOK;
    __shared__ float red[128];
    const int t = threadIdx.x;

    float v[8];
    float mx = -1e30f;
    #pragma unroll
    for (int i = 0; i < 8; ++i) { v[i] = s[t + i * 128]; mx = fmaxf(mx, v[i]); }
    red[t] = mx; __syncthreads();
    for (int off = 64; off > 0; off >>= 1) {
        if (t < off) red[t] = fmaxf(red[t], red[t + off]);
        __syncthreads();
    }
    mx = red[0]; __syncthreads();

    float sum = 0.0f;
    #pragma unroll
    for (int i = 0; i < 8; ++i) { v[i] = __expf(v[i] - mx); sum += v[i]; }
    red[t] = sum; __syncthreads();
    for (int off = 64; off > 0; off >>= 1) {
        if (t < off) red[t] += red[t + off];
        __syncthreads();
    }
    const float inv = 1.0f / red[0];
    #pragma unroll
    for (int i = 0; i < 8; ++i) p[t + i * 128] = (_Float16)(v[i] * inv);
}

// x = LN(x + resid) * s + b ; writes f32 and f16. One block (128 thr) per row.
__global__ void add_ln_kernel(float* __restrict__ x,
                              const float* __restrict__ resid,
                              const float* __restrict__ s,
                              const float* __restrict__ b,
                              _Float16* __restrict__ xh)
{
    const int row = blockIdx.x, t = threadIdx.x;
    __shared__ float red[128];
    const float v = x[row * DM + t] + resid[row * DM + t];
    red[t] = v; __syncthreads();
    for (int off = 64; off > 0; off >>= 1) {
        if (t < off) red[t] += red[t + off];
        __syncthreads();
    }
    const float mean = red[0] * (1.0f / DM); __syncthreads();
    const float d = v - mean;
    red[t] = d * d; __syncthreads();
    for (int off = 64; off > 0; off >>= 1) {
        if (t < off) red[t] += red[t + off];
        __syncthreads();
    }
    const float var = red[0] * (1.0f / DM);
    const float y = d * rsqrtf(var + LN_EPS) * s[t] + b[t];
    x[row * DM + t]  = y;
    xh[row * DM + t] = (_Float16)y;
}

// out[i][j] = relu(a[i,:] + b[j,:] + b1) . w2 + b2 ; 16x16 tile per block
__global__ void pair_kernel(const float* __restrict__ A,
                            const float* __restrict__ B,
                            const float* __restrict__ b1,
                            const float* __restrict__ w2,
                            const float* __restrict__ b2,
                            float* __restrict__ out)
{
    __shared__ float sa[16][132];   // padded: stride%64 == 4 banks -> no conflict
    __shared__ float sb[16][132];
    __shared__ float sw[128], s1[128];
    const int i0 = blockIdx.y * 16, j0 = blockIdx.x * 16;
    const int tid = threadIdx.y * 16 + threadIdx.x;
    for (int idx = tid; idx < 16 * 128; idx += 256) {
        const int r = idx >> 7, c = idx & 127;
        sa[r][c] = A[(size_t)(i0 + r) * DM + c];
        sb[r][c] = B[(size_t)(j0 + r) * DM + c];
    }
    if (tid < 128) { sw[tid] = w2[tid]; s1[tid] = b1[tid]; }
    __syncthreads();

    const int ti = threadIdx.y, tj = threadIdx.x;
    float acc = b2[0];
    #pragma unroll 8
    for (int k = 0; k < 128; ++k) {
        float h = sa[ti][k] + sb[tj][k] + s1[k];
        h = fmaxf(h, 0.0f);
        acc = fmaf(h, sw[k], acc);
    }
    out[(size_t)(i0 + ti) * TOK + (j0 + tj)] = acc;
}

// ---------------------------------------------------------------------------
extern "C" void kernel_launch(void* const* d_in, const int* in_sizes, int n_in,
                              void* d_out, int out_size, void* d_ws, size_t ws_size,
                              hipStream_t stream)
{
    const int*   ids    = (const int*)  d_in[0];
    const float* emb    = (const float*)d_in[1];
    const float* qkv_w  = (const float*)d_in[2];   // (2,384,128)
    const float* qkv_b  = (const float*)d_in[3];   // (2,384)
    const float* ow     = (const float*)d_in[4];   // (2,128,128)
    const float* ob     = (const float*)d_in[5];
    const float* ln1_s  = (const float*)d_in[6];
    const float* ln1_b  = (const float*)d_in[7];
    const float* ln2_s  = (const float*)d_in[8];
    const float* ln2_b  = (const float*)d_in[9];
    const float* ff1_w  = (const float*)d_in[10];  // (2,2048,128)
    const float* ff1_b  = (const float*)d_in[11];
    const float* ff2_w  = (const float*)d_in[12];  // (2,128,2048)
    const float* ff2_b  = (const float*)d_in[13];
    const float* w1     = (const float*)d_in[14];  // (128,256)
    const float* b1     = (const float*)d_in[15];
    const float* w2     = (const float*)d_in[16];  // (1,128)
    const float* b2     = (const float*)d_in[17];

    // workspace carve-up (all chunks 256B aligned)
    char* p = (char*)d_ws;
    auto carve = [&](size_t bytes) -> void* {
        void* r = (void*)p;
        p += (bytes + 255) & ~(size_t)255;
        return r;
    };
    float*    x_f32   = (float*)   carve((size_t)TOK * DM * 4);
    float*    proj    = (float*)   carve((size_t)TOK * DM * 4);
    float*    ff2o    = (float*)   carve((size_t)TOK * DM * 4);
    float*    a_f32   = (float*)   carve((size_t)TOK * DM * 4);
    float*    b_f32   = (float*)   carve((size_t)TOK * DM * 4);
    float*    scores  = (float*)   carve((size_t)NHEADS * TOK * TOK * 4);
    _Float16* x_h     = (_Float16*)carve((size_t)TOK * DM * 2);
    _Float16* qkv_h   = (_Float16*)carve((size_t)TOK * 3 * DM * 2);
    _Float16* vt_h    = (_Float16*)carve((size_t)NHEADS * HDIM * TOK * 2);
    _Float16* attn_h  = (_Float16*)carve((size_t)TOK * DM * 2);
    _Float16* ff_h    = (_Float16*)carve((size_t)TOK * FFD * 2);
    _Float16* probs_h = (_Float16*)carve((size_t)NHEADS * TOK * TOK * 2);
    _Float16* qkvw_h  = (_Float16*)carve((size_t)2 * 3 * DM * DM * 2);
    _Float16* ow_h    = (_Float16*)carve((size_t)2 * DM * DM * 2);
    _Float16* ff1w_h  = (_Float16*)carve((size_t)2 * FFD * DM * 2);
    _Float16* ff2w_h  = (_Float16*)carve((size_t)2 * DM * FFD * 2);
    _Float16* w1_h    = (_Float16*)carve((size_t)DM * 2 * DM * 2);

    auto cvt = [&](const float* s, _Float16* d, int n) {
        f32_to_f16_kernel<<<(n + 255) / 256, 256, 0, stream>>>(s, d, n);
    };
    cvt(qkv_w, qkvw_h, 2 * 3 * DM * DM);
    cvt(ow,    ow_h,   2 * DM * DM);
    cvt(ff1_w, ff1w_h, 2 * FFD * DM);
    cvt(ff2_w, ff2w_h, 2 * DM * FFD);
    cvt(w1,    w1_h,   DM * 2 * DM);

    embed_kernel<<<TOK, DM, 0, stream>>>(ids, emb, x_f32, x_h);

    const float inv_sqrt_hd = 0.17677669529663687f;   // 1/sqrt(32)

    for (int l = 0; l < 2; ++l) {
        const _Float16* Wqkv = qkvw_h + (size_t)l * 3 * DM * DM;
        const _Float16* Wo   = ow_h   + (size_t)l * DM * DM;
        const _Float16* W1f  = ff1w_h + (size_t)l * FFD * DM;
        const _Float16* W2f  = ff2w_h + (size_t)l * DM * FFD;

        // QKV: (1024x384) = x @ Wqkv^T + b
        gemm_wmma_nt<4><<<dim3(6, 16, 1), 128, 0, stream>>>(
            x_h, DM, 0, Wqkv, DM, 0, qkv_b + l * 3 * DM,
            nullptr, qkv_h, 3 * DM, 0, TOK, 3 * DM, DM, 1.0f, 0);

        build_vt_kernel<<<(NHEADS * HDIM * TOK) / 256, 256, 0, stream>>>(qkv_h, vt_h);

        // scores[h] = Q_h @ K_h^T / sqrt(hd)   (K=32, per-head column offsets)
        gemm_wmma_nt<4><<<dim3(16, 16, NHEADS), 128, 0, stream>>>(
            qkv_h, 3 * DM, HDIM, qkv_h + DM, 3 * DM, HDIM, nullptr,
            scores, nullptr, TOK, (long)TOK * TOK,
            TOK, TOK, HDIM, inv_sqrt_hd, 0);

        softmax_kernel<<<NHEADS * TOK, 128, 0, stream>>>(scores, probs_h);

        // O_h = P_h @ V_h  (B = V^T, contiguous rows) -> attn_h columns h*32..
        gemm_wmma_nt<2><<<dim3(1, 16, NHEADS), 128, 0, stream>>>(
            probs_h, TOK, (long)TOK * TOK, vt_h, TOK, (long)HDIM * TOK, nullptr,
            nullptr, attn_h, DM, HDIM, TOK, HDIM, TOK, 1.0f, 0);

        // attention output projection
        gemm_wmma_nt<4><<<dim3(2, 16, 1), 128, 0, stream>>>(
            attn_h, DM, 0, Wo, DM, 0, ob + l * DM,
            proj, nullptr, DM, 0, TOK, DM, DM, 1.0f, 0);

        add_ln_kernel<<<TOK, DM, 0, stream>>>(x_f32, proj,
                                              ln1_s + l * DM, ln1_b + l * DM, x_h);

        // FFN
        gemm_wmma_nt<4><<<dim3(32, 16, 1), 128, 0, stream>>>(
            x_h, DM, 0, W1f, DM, 0, ff1_b + l * FFD,
            nullptr, ff_h, FFD, 0, TOK, FFD, DM, 1.0f, 1);

        gemm_wmma_nt<4><<<dim3(2, 16, 1), 128, 0, stream>>>(
            ff_h, FFD, 0, W2f, FFD, 0, ff2_b + l * DM,
            ff2o, nullptr, DM, 0, TOK, DM, FFD, 1.0f, 0);

        add_ln_kernel<<<TOK, DM, 0, stream>>>(x_f32, ff2o,
                                              ln2_s + l * DM, ln2_b + l * DM, x_h);
    }

    // pair MLP halves: a = x @ w1[:, :128]^T ; b = x @ w1[:, 128:]^T
    gemm_wmma_nt<4><<<dim3(2, 16, 1), 128, 0, stream>>>(
        x_h, DM, 0, w1_h, 2 * DM, 0, nullptr,
        a_f32, nullptr, DM, 0, TOK, DM, DM, 1.0f, 0);
    gemm_wmma_nt<4><<<dim3(2, 16, 1), 128, 0, stream>>>(
        x_h, DM, 0, w1_h + DM, 2 * DM, 0, nullptr,
        b_f32, nullptr, DM, 0, TOK, DM, DM, 1.0f, 0);

    pair_kernel<<<dim3(TOK / 16, TOK / 16), dim3(16, 16), 0, stream>>>(
        a_f32, b_f32, b1, w2, b2, (float*)d_out);
}
